// GNNLinkPredictor_41781441855493
// MI455X (gfx1250) — matrix operand order repaired and checked by reading, not compiled, not run on hardware
//
#include <hip/hip_runtime.h>

// ---------------------------------------------------------------------------
// GCN link predictor on gfx1250 (MI455X, wave32, WMMA).
//   h  = relu( Âhat (X W1) + b1 )        GEMM via v_wmma_f32_16x16x32_bf16
//   z  = Âhat (h W2) + b2
//   out[k] = dot(z[src_k], z[dst_k])
// Âhat = D^-1/2 (A + I) D^-1/2 applied as: self-loop init + edge atomic scatter.
// ---------------------------------------------------------------------------

typedef __attribute__((ext_vector_type(16))) __bf16 v16bf;
typedef __attribute__((ext_vector_type(8)))  __bf16 v8bf;
typedef __attribute__((ext_vector_type(8)))  float  v8f;

#define HID 256

// ---- degree / normalization ------------------------------------------------
__global__ void k_fill_ones(float* p, int n) {
    int i = blockIdx.x * blockDim.x + threadIdx.x;
    if (i < n) p[i] = 1.0f;                       // self-loop contributes 1
}

__global__ void k_deg_accum(const int* __restrict__ col, float* __restrict__ deg, int e) {
    int i = blockIdx.x * blockDim.x + threadIdx.x;
    if (i < e) atomicAdd(&deg[col[i]], 1.0f);
}

__global__ void k_rsqrt_inplace(float* p, int n) {
    int i = blockIdx.x * blockDim.x + threadIdx.x;
    if (i < n) p[i] = rsqrtf(p[i]);               // deg >= 1 always (self-loop)
}

// ---- dtype staging ---------------------------------------------------------
__global__ void k_f32_to_bf16(const float* __restrict__ a, __bf16* __restrict__ b, int n) {
    int i = blockIdx.x * blockDim.x + threadIdx.x;
    if (i < n) b[i] = (__bf16)a[i];
}

// W[k, c] (row-major K x C) -> Wt[c, k] bf16, so a WMMA B-fragment is a
// contiguous 32B run of K per lane.
__global__ void k_w_transpose_bf16(const float* __restrict__ w, __bf16* __restrict__ wt,
                                   int K, int C) {
    int i = blockIdx.x * blockDim.x + threadIdx.x;
    if (i < K * C) {
        int k = i / C, c = i % C;
        wt[(size_t)c * K + k] = (__bf16)w[i];
    }
}

// ---- GEMM: C[M,Ncols] = A[M,K](bf16) * Bt[Ncols,K](bf16), f32 accumulate ---
// One wave computes one 16x16 tile. Requires M%16==0, Ncols%16==0, K%32==0.
__global__ void k_gemm_bf16_wmma(const __bf16* __restrict__ A, const __bf16* __restrict__ Bt,
                                 float* __restrict__ C, int M, int Ncols, int K) {
    const int lane    = threadIdx.x & 31;
    const int wave    = threadIdx.x >> 5;
    const int tiles_n = Ncols >> 4;
    const int tile    = blockIdx.x * (blockDim.x >> 5) + wave;
    if (tile >= (M >> 4) * tiles_n) return;       // wave-uniform: EXEC stays all-1s
    const int tm = tile / tiles_n;
    const int tn = tile % tiles_n;
    const int hg = lane >> 4;                     // half-wave group (K split)
    const int lr = lane & 15;                     // row (A) / column (B) within tile

    const __bf16* ap = A  + (size_t)(tm * 16 + lr) * K + 16 * hg;
    const __bf16* bp = Bt + (size_t)(tn * 16 + lr) * K + 16 * hg;

    v8f acc = {};
    for (int k0 = 0; k0 < K; k0 += 32) {
        v8bf alo = *(const v8bf*)(ap + k0);       // K = base .. base+7
        v8bf ahi = *(const v8bf*)(ap + k0 + 8);   // K = base+8 .. base+15
        v16bf a, b;
        b = *(const v16bf*)(bp + k0);             // 16 contiguous K values
#pragma unroll
        for (int i = 0; i < 8; ++i) { a[i] = alo[i]; a[i + 8] = ahi[i]; }
        acc = __builtin_amdgcn_wmma_f32_16x16x32_bf16(
            /*neg_a=*/false, a, /*neg_b=*/false, b,
            /*c_mod=*/(short)0, acc, /*reuse_a=*/false, /*reuse_b=*/false);
    }
    // D layout: VGPR r -> row (r + 8*hg), column lr
    float* cp = C + (size_t)(tm * 16 + 8 * hg) * Ncols + tn * 16 + lr;
#pragma unroll
    for (int r = 0; r < 8; ++r) cp[(size_t)r * Ncols] = acc[r];
}

// ---- propagate: agg = selfloop(h) + scatter(edges) -------------------------
__global__ void k_selfloop_init(const float* __restrict__ h, const float* __restrict__ dinv,
                                float* __restrict__ agg, int total) {
    int i = blockIdx.x * blockDim.x + threadIdx.x;
    if (i < total) {
        float s = dinv[i >> 8];                   // HID == 256
        agg[i] = h[i] * s * s;
    }
}

// One wave per edge: coalesced 128B gather per channel-group, f32 atomics to L2.
__global__ void k_edge_scatter(const float* __restrict__ h, const int* __restrict__ row,
                               const int* __restrict__ col, const float* __restrict__ dinv,
                               float* __restrict__ agg, int E) {
    const int lane = threadIdx.x & 31;
    const int w    = blockIdx.x * (blockDim.x >> 5) + (threadIdx.x >> 5);
    if (w >= E) return;
    const int r = row[w], c = col[w];
    const float nrm = dinv[r] * dinv[c];
    const float* hp = h + (size_t)r * HID;
    float* ap = agg + (size_t)c * HID;
#pragma unroll
    for (int j = 0; j < 8; ++j) {
        const int ch = lane + 32 * j;
        atomicAdd(ap + ch, hp[ch] * nrm);
    }
}

__global__ void k_bias_relu_bf16(const float* __restrict__ agg, const float* __restrict__ b,
                                 __bf16* __restrict__ o, int total) {
    int i = blockIdx.x * blockDim.x + threadIdx.x;
    if (i < total) {
        float v = agg[i] + b[i & (HID - 1)];
        o[i] = (__bf16)fmaxf(v, 0.0f);
    }
}

__global__ void k_bias_add(float* __restrict__ agg, const float* __restrict__ b, int total) {
    int i = blockIdx.x * blockDim.x + threadIdx.x;
    if (i < total) agg[i] += b[i & (HID - 1)];
}

// ---- decode: one wave per label edge, 256-dim dot + shfl_xor reduction -----
__global__ void k_decode(const float* __restrict__ z, const int* __restrict__ s,
                         const int* __restrict__ d, float* __restrict__ o, int L) {
    const int lane = threadIdx.x & 31;
    const int w    = blockIdx.x * (blockDim.x >> 5) + (threadIdx.x >> 5);
    if (w >= L) return;
    const float* ps = z + (size_t)s[w] * HID + lane * 8;
    const float* pd = z + (size_t)d[w] * HID + lane * 8;
    float acc = 0.0f;
#pragma unroll
    for (int j = 0; j < 8; ++j) acc += ps[j] * pd[j];
#pragma unroll
    for (int off = 16; off; off >>= 1) acc += __shfl_xor(acc, off, 32);
    if (lane == 0) o[w] = acc;
}

// ---------------------------------------------------------------------------
extern "C" void kernel_launch(void* const* d_in, const int* in_sizes, int n_in,
                              void* d_out, int out_size, void* d_ws, size_t ws_size,
                              hipStream_t stream) {
    const float* x   = (const float*)d_in[0];
    const int*   ei  = (const int*)d_in[1];   // [2, E] : row = ei, col = ei + E
    const int*   eli = (const int*)d_in[2];   // [2, L]
    const float* W1  = (const float*)d_in[3];
    const float* b1  = (const float*)d_in[4];
    const float* W2  = (const float*)d_in[5];
    const float* b2  = (const float*)d_in[6];
    float* out = (float*)d_out;

    const int IN_CH = 128;
    const int N = in_sizes[0] / IN_CH;        // 50000 (multiple of 16)
    const int E = in_sizes[1] / 2;            // 800000
    const int L = in_sizes[2] / 2;            // 200000

    // ---- carve workspace (all buffers fully written before read) ----
    char* base = (char*)d_ws;
    size_t off = 0;
    auto carve = [&](size_t bytes) -> void* {
        void* r = base + off;
        off += (bytes + 255) & ~(size_t)255;
        return r;
    };
    float*  dinv = (float*)carve((size_t)N * 4);                 // deg -> dinv in place
    __bf16* xbf  = (__bf16*)carve((size_t)N * IN_CH * 2);
    __bf16* w1t  = (__bf16*)carve((size_t)IN_CH * HID * 2);
    __bf16* w2t  = (__bf16*)carve((size_t)HID * HID * 2);
    float*  h    = (float*)carve((size_t)N * HID * 4);           // GEMM out (reused for z)
    float*  agg  = (float*)carve((size_t)N * HID * 4);           // propagate out (reused)
    __bf16* hbf  = (__bf16*)carve((size_t)N * HID * 2);
    (void)ws_size; (void)n_in; (void)out_size;

    const int TPB = 256;
    const int NH  = N * HID;

    // 1) symmetric normalization: deg (with self-loops) -> d^{-1/2}
    k_fill_ones<<<(N + TPB - 1) / TPB, TPB, 0, stream>>>(dinv, N);
    k_deg_accum<<<(E + TPB - 1) / TPB, TPB, 0, stream>>>(ei + E, dinv, E);
    k_rsqrt_inplace<<<(N + TPB - 1) / TPB, TPB, 0, stream>>>(dinv, N);

    // 2) stage bf16 operands (weights transposed for WMMA B fragments)
    k_f32_to_bf16<<<((size_t)N * IN_CH + TPB - 1) / TPB, TPB, 0, stream>>>(x, xbf, N * IN_CH);
    k_w_transpose_bf16<<<(IN_CH * HID + TPB - 1) / TPB, TPB, 0, stream>>>(W1, w1t, IN_CH, HID);
    k_w_transpose_bf16<<<(HID * HID + TPB - 1) / TPB, TPB, 0, stream>>>(W2, w2t, HID, HID);

    // 3) layer 1: h = X W1 (WMMA), propagate, relu -> bf16
    {
        int tiles = (N / 16) * (HID / 16);
        k_gemm_bf16_wmma<<<(tiles + 3) / 4, 128, 0, stream>>>(xbf, w1t, h, N, HID, IN_CH);
    }
    k_selfloop_init<<<(NH + TPB - 1) / TPB, TPB, 0, stream>>>(h, dinv, agg, NH);
    k_edge_scatter<<<(E + 7) / 8, TPB, 0, stream>>>(h, ei, ei + E, dinv, agg, E);
    k_bias_relu_bf16<<<(NH + TPB - 1) / TPB, TPB, 0, stream>>>(agg, b1, hbf, NH);

    // 4) layer 2: z = h W2 (WMMA, into h buffer), propagate (into agg), + b2
    {
        int tiles = (N / 16) * (HID / 16);
        k_gemm_bf16_wmma<<<(tiles + 3) / 4, 128, 0, stream>>>(hbf, w2t, h, N, HID, HID);
    }
    k_selfloop_init<<<(NH + TPB - 1) / TPB, TPB, 0, stream>>>(h, dinv, agg, NH);
    k_edge_scatter<<<(E + 7) / 8, TPB, 0, stream>>>(h, ei, ei + E, dinv, agg, E);
    k_bias_add<<<(NH + TPB - 1) / TPB, TPB, 0, stream>>>(agg, b2, NH);

    // 5) edge decode
    k_decode<<<(L + 7) / 8, TPB, 0, stream>>>(agg, eli, eli + L, out, L);
}
